// CSNeuralODE_23871428231356
// MI455X (gfx1250) — compile-verified, matrix operand order checked by reading, not run on hardware
//
#include <hip/hip_runtime.h>
#include <hip/hip_bf16.h>

// ---------------------------------------------------------------------------
// CDNA5 (gfx1250) Neural-ODE Euler integrator.
//   y_{s+1} = y_s + ( f(y_s) + g(y_s) * sin(t_s * freqs) ) * dt
// f: 64->256 tanh ->256 softplus ->256 softplus ->64
// g: 64->52 tanh ->64            (52 padded to 64 with zeros)
// Matmuls: v_wmma_f32_16x16x32_f16 (f16 in, f32 accumulate).
// Activations: hardware TRANS32 ops (v_exp/v_log/v_sin/v_rcp) to keep the
// epilogue branch-free and co-executable with the WMMA pipe.
// ---------------------------------------------------------------------------

typedef __attribute__((ext_vector_type(16))) _Float16 v16h;
typedef __attribute__((ext_vector_type(8)))  _Float16 v8h;
typedef __attribute__((ext_vector_type(8)))  float    v8f;

union V16 { v16h v; v8h h[2]; };

// Load a 16-half WMMA fragment given this lane's base pointer.
// Per CDNA5 16-bit fragment layout the lane's 16 values are two contiguous
// 8-half (16B) chunks separated by 16 K-elements.
__device__ __forceinline__ v16h ld_frag16(const _Float16* p) {
  V16 u;
  u.h[0] = *(const v8h*)(p);
  u.h[1] = *(const v8h*)(p + 16);
  return u.v;
}

// Branch-free fast activations on TRANS32 hardware ops.
__device__ __forceinline__ float fast_tanh(float x) {
  float t = __expf(-2.0f * __builtin_fabsf(x));          // v_exp_f32
  float r = (1.0f - t) * __builtin_amdgcn_rcpf(1.0f + t); // v_rcp_f32
  return __builtin_copysignf(r, x);
}
__device__ __forceinline__ float fast_softplus(float x) {
  // max(x,0) + log(1 + exp(-|x|)) : numerically stable softplus
  return __builtin_fmaxf(x, 0.0f) +
         __logf(1.0f + __expf(-__builtin_fabsf(x)));      // v_log_f32, v_exp_f32
}

// ---------------------------------------------------------------------------
// Prep: fp32 row-major weight [K,N] -> f16 column-major [Np][Kp], zero-padded.
// Column-major makes each WMMA B-fragment chunk contiguous per lane.
// ---------------------------------------------------------------------------
__global__ void convert_w(const float* __restrict__ src, _Float16* __restrict__ dst,
                          int K, int N, int Kp, int Np) {
  int i = blockIdx.x * blockDim.x + threadIdx.x;
  if (i >= Kp * Np) return;
  int n = i / Kp;
  int k = i - n * Kp;
  float v = (k < K && n < N) ? src[k * N + n] : 0.0f;
  dst[n * Kp + k] = (_Float16)v;
}

// Prep: zero-padded f32 bias copy (removes divergent guarded loads in GEMM).
__global__ void convert_b(const float* __restrict__ src, float* __restrict__ dst,
                          int N, int Np) {
  int i = blockIdx.x * blockDim.x + threadIdx.x;
  if (i >= Np) return;
  dst[i] = (i < N) ? src[i] : 0.0f;
}

// ---------------------------------------------------------------------------
// One GEMM layer:  out = act( A[32,K] * W[K,N] + bias ),  W col-major f16.
// 8 waves, each owns (2*N/16)/8 16x16 tiles.  ACT: 0=tanh, 1=softplus, 2=none.
// ---------------------------------------------------------------------------
template<int K, int N, int ACT, bool F32OUT>
__device__ __forceinline__ void gemm_layer(const _Float16* __restrict__ A, int lda,
                                           const _Float16* __restrict__ Wt,
                                           const float*    __restrict__ bias,
                                           _Float16* outH, float* outF, int ldo) {
  const int lane = threadIdx.x & 31;
  const int wave = threadIdx.x >> 5;
  const int lm   = lane & 15;          // col within tile (B/C/D), row for A
  const int kh   = (lane >> 4) << 3;   // 0 or 8: K-chunk select / M-half select
  const int NT   = N >> 4;
  const int TT   = 2 * NT;             // M=32 -> 2 row tiles

  for (int tile = wave; tile < TT; tile += 8) {
    const int m0 = (tile & 1) << 4;
    const int n0 = (tile >> 1) << 4;
    v8f acc = {};
#pragma unroll
    for (int k0 = 0; k0 < K; k0 += 32) {
      v16h a = ld_frag16(A  + (m0 + lm) * lda + k0 + kh);   // LDS, row-major
      v16h b = ld_frag16(Wt + (n0 + lm) * K   + k0 + kh);   // global, col-major
      acc = __builtin_amdgcn_wmma_f32_16x16x32_f16(
                false, a, false, b, (short)0, acc, false, false);
    }
    const int   n  = n0 + lm;
    const float bv = bias[n];          // padded: uniform, branch-free
    const int   mb = m0 + kh;          // D rows: m0 + i + 8*(lane>=16)
#pragma unroll
    for (int i = 0; i < 8; ++i) {
      float v = acc[i] + bv;
      if (ACT == 0)      v = fast_tanh(v);
      else if (ACT == 1) v = fast_softplus(v);
      if (F32OUT) outF[(mb + i) * ldo + n] = v;
      else        outH[(mb + i) * ldo + n] = (_Float16)v;
    }
  }
}

// ---------------------------------------------------------------------------
// Main integrator: one workgroup integrates a 32-row batch tile through all
// time steps, keeping state in LDS (weights stream from L2 every step).
// ---------------------------------------------------------------------------
__global__ __launch_bounds__(256)
void ode_kernel(const float* __restrict__ y0, const float* __restrict__ tg,
                const _Float16* __restrict__ Wf0t, const float* __restrict__ bf0,
                const _Float16* __restrict__ Wf1t, const float* __restrict__ bf1,
                const _Float16* __restrict__ Wf2t, const float* __restrict__ bf2,
                const _Float16* __restrict__ Wf3t, const float* __restrict__ bf3,
                const _Float16* __restrict__ Wg0t, const float* __restrict__ bg0,
                const _Float16* __restrict__ Wg1t, const float* __restrict__ bg1,
                float* __restrict__ out, int nstep) {
  __shared__ float    y_f[32 * 64];     // master fp32 state
  __shared__ _Float16 y_h[32 * 64];     // f16 mirror (WMMA A operand)
  __shared__ _Float16 hA [32 * 256];    // activation ping
  __shared__ _Float16 hB [32 * 256];    // activation pong
  __shared__ _Float16 hg [32 * 64];     // g hidden (padded 52->64)
  __shared__ float    f_l[32 * 64];     // f branch result
  __shared__ float    g_l[32 * 64];     // g branch result
  __shared__ float    u_l[64];          // control signal sin(t*freqs)

  const int tid = threadIdx.x;
  const int b0  = blockIdx.x * 32;

  // init state
  for (int e = tid; e < 32 * 64; e += 256) {
    float v = y0[b0 * 64 + e];
    y_f[e] = v;
    y_h[e] = (_Float16)v;
  }
  const float dt = tg[1] - tg[0];
  __syncthreads();

  for (int s = 0; s < nstep; ++s) {
    const float ti = tg[s];
    if (tid < 64) u_l[tid] = __sinf(ti * (float)(tid + 1));  // freqs = 1..64

    // g hidden: tanh(y @ Wg0 + bg0)            [32,64]x[64,64]
    gemm_layer<64, 64, 0, false>(y_h, 64, Wg0t, bg0, hg, nullptr, 64);
    // f layer0: tanh(y @ Wf0 + bf0)            [32,64]x[64,256]
    gemm_layer<64, 256, 0, false>(y_h, 64, Wf0t, bf0, hA, nullptr, 256);
    __syncthreads();
    // f layer1: softplus(hA @ Wf1 + bf1)       [32,256]x[256,256]
    gemm_layer<256, 256, 1, false>(hA, 256, Wf1t, bf1, hB, nullptr, 256);
    __syncthreads();
    // f layer2: softplus(hB @ Wf2 + bf2)
    gemm_layer<256, 256, 1, false>(hB, 256, Wf2t, bf2, hA, nullptr, 256);
    __syncthreads();
    // f out:    hA @ Wf3 + bf3                 [32,256]x[256,64] -> f32
    gemm_layer<256, 64, 2, true>(hA, 256, Wf3t, bf3, nullptr, f_l, 64);
    // g out:    hg @ Wg1 + bg1                 [32,64]x[64,64]   -> f32
    gemm_layer<64, 64, 2, true>(hg, 64, Wg1t, bg1, nullptr, g_l, 64);
    __syncthreads();

    // Euler update: y += (f + g*u) * dt
    for (int e = tid; e < 32 * 64; e += 256) {
      const int d = e & 63;
      float v = y_f[e] + (f_l[e] + g_l[e] * u_l[d]) * dt;
      y_f[e] = v;
      y_h[e] = (_Float16)v;
    }
    __syncthreads();
  }

  for (int e = tid; e < 32 * 64; e += 256)
    out[b0 * 64 + e] = y_f[e];
}

// ---------------------------------------------------------------------------
extern "C" void kernel_launch(void* const* d_in, const int* in_sizes, int n_in,
                              void* d_out, int out_size, void* d_ws, size_t ws_size,
                              hipStream_t stream) {
  const float* y0  = (const float*)d_in[0];
  const float* t   = (const float*)d_in[1];
  const float* Wf0 = (const float*)d_in[2];  const float* bf0 = (const float*)d_in[3];
  const float* Wf1 = (const float*)d_in[4];  const float* bf1 = (const float*)d_in[5];
  const float* Wf2 = (const float*)d_in[6];  const float* bf2 = (const float*)d_in[7];
  const float* Wf3 = (const float*)d_in[8];  const float* bf3 = (const float*)d_in[9];
  const float* Wg0 = (const float*)d_in[10]; const float* bg0 = (const float*)d_in[11];
  const float* Wg1 = (const float*)d_in[12]; const float* bg1 = (const float*)d_in[13];

  // f16 col-major weight staging + padded f32 biases in workspace (~348 KB)
  _Float16* ws   = (_Float16*)d_ws;
  _Float16* Wf0t = ws;                    // [256][64]
  _Float16* Wf1t = Wf0t + 256 * 64;       // [256][256]
  _Float16* Wf2t = Wf1t + 256 * 256;      // [256][256]
  _Float16* Wf3t = Wf2t + 256 * 256;      // [64][256]
  _Float16* Wg0t = Wf3t + 64 * 256;       // [64][64] (N padded 52->64)
  _Float16* Wg1t = Wg0t + 64 * 64;        // [64][64] (K padded 52->64)
  float*    bptr = (float*)(Wg1t + 64 * 64);
  float* bf0p = bptr;        // 256
  float* bf1p = bf0p + 256;  // 256
  float* bf2p = bf1p + 256;  // 256
  float* bf3p = bf2p + 256;  // 64
  float* bg0p = bf3p + 64;   // 64 (padded 52->64, zeros)
  float* bg1p = bg0p + 64;   // 64

  auto cvtw = [&](const float* src, _Float16* dst, int K, int N, int Kp, int Np) {
    int cnt = Kp * Np;
    convert_w<<<(cnt + 255) / 256, 256, 0, stream>>>(src, dst, K, N, Kp, Np);
  };
  cvtw(Wf0, Wf0t, 64, 256, 64, 256);
  cvtw(Wf1, Wf1t, 256, 256, 256, 256);
  cvtw(Wf2, Wf2t, 256, 256, 256, 256);
  cvtw(Wf3, Wf3t, 256, 64, 256, 64);
  cvtw(Wg0, Wg0t, 64, 52, 64, 64);
  cvtw(Wg1, Wg1t, 52, 64, 64, 64);

  auto cvtb = [&](const float* src, float* dst, int N, int Np) {
    convert_b<<<(Np + 255) / 256, 256, 0, stream>>>(src, dst, N, Np);
  };
  cvtb(bf0, bf0p, 256, 256);
  cvtb(bf1, bf1p, 256, 256);
  cvtb(bf2, bf2p, 256, 256);
  cvtb(bf3, bf3p, 64, 64);
  cvtb(bg0, bg0p, 52, 64);
  cvtb(bg1, bg1p, 64, 64);

  const int B     = in_sizes[0] / 64;   // 4096
  const int T     = in_sizes[1];        // 200
  const int nstep = T - 1;              // 199 Euler steps

  ode_kernel<<<B / 32, 256, 0, stream>>>(
      y0, t, Wf0t, bf0p, Wf1t, bf1p, Wf2t, bf2p, Wf3t, bf3p,
      Wg0t, bg0p, Wg1t, bg1p, (float*)d_out, nstep);
}
